// BlockMEConcat_39307540693473
// MI455X (gfx1250) — matrix operand order repaired and checked by reading, not compiled, not run on hardware
//
#include <hip/hip_runtime.h>
#include <cstdint>
#include <cstddef>

// ---------------- problem constants (match reference) ----------------
#define NPTS   1000000
#define KOFF   27
#define CIN    64
#define COUT   64
#define TILES  (NPTS / 16)      // 62500 row-tiles of 16 points

// ---------------- CDNA5 WMMA / vector types ----------------
typedef __attribute__((ext_vector_type(16))) __bf16       v16bf;
typedef __attribute__((ext_vector_type(8)))  float        v8f;
typedef __attribute__((ext_vector_type(4)))  unsigned int u32v4;
typedef __attribute__((ext_vector_type(4)))  float        f32x4;

union BfFrag {                  // 32 bytes = 8 VGPRs = one 16x32 bf16 A/B fragment per lane
  u32v4 u[2];
  v16bf v;
};

#define WMMA_BF16(A, B, C) \
  __builtin_amdgcn_wmma_f32_16x16x32_bf16(false, (A), false, (B), (short)0, (C), false, false)

__device__ __forceinline__ unsigned short f2bf(float f) {
  unsigned int u = __float_as_uint(f);
  unsigned int r = (u + 0x7FFFu + ((u >> 16) & 1u)) >> 16;   // round-to-nearest-even
  return (unsigned short)r;
}

// ---------------- kernel 1: x = concat(feats, time_emb[batch_idx]) as bf16 [N][64] ----------------
// x_bf (128 MB) is the gather target of the conv: it must become and stay L2-resident (192 MB L2).
// feats / batch_idx are streamed exactly once -> non-temporal loads so they don't pollute L2.
__global__ __launch_bounds__(256)
void prep_x(const float* __restrict__ feats, const float* __restrict__ temb,
            const int* __restrict__ bidx, unsigned short* __restrict__ xbf)
{
  size_t t = (size_t)blockIdx.x * 256 + threadIdx.x;   // N*16 threads, 4 channels each
  int n  = (int)(t >> 4);
  int c4 = (int)(t & 15) << 2;
  f32x4 v;
  if (c4 < 32) {
    v = __builtin_nontemporal_load((const f32x4*)(feats + (size_t)n * 32 + c4));
  } else {
    int b = __builtin_nontemporal_load(bidx + n);
    v = *(const f32x4*)(temb + (size_t)b * 32 + (c4 - 32));   // temb is tiny & reused: keep cached
  }
  unsigned long long p = (unsigned long long)f2bf(v.x)
                       | ((unsigned long long)f2bf(v.y) << 16)
                       | ((unsigned long long)f2bf(v.z) << 32)
                       | ((unsigned long long)f2bf(v.w) << 48);
  *(unsigned long long*)(xbf + (size_t)n * 64 + c4) = p;      // regular store: want it in L2
}

// ---------------- kernel 2: swizzle W[k][cin][cout] f32 -> per-lane bf16 B fragments --------------
// Block b = ((k*2 + kchunk)*4 + coltile); fragment = 32 lanes x 16 bf16 (32B/lane, contiguous 1KB).
// ISA B layout (16-bit, 32x16): lanes 0-15: elems 0-7 K=0..7, 8-15 K=16..23; lanes 16-31: K=8..15, 24..31;
// column N = lane % 16.
__global__ __launch_bounds__(256)
void prep_w(const float* __restrict__ W, unsigned short* __restrict__ wfrag)
{
  int t = blockIdx.x * 256 + threadIdx.x;              // 27*8*512 = 110592 threads
  int b  = t >> 9;
  int l  = (t >> 4) & 31;
  int e  = t & 15;
  int k     = b >> 3;
  int chunk = (b >> 2) & 1;
  int ct    = b & 3;
  int cin  = chunk * 32 + ((e < 8) ? ((l >> 4) * 8 + e) : (16 + (l >> 4) * 8 + (e - 8)));
  int cout = ct * 16 + (l & 15);
  wfrag[t] = f2bf(W[((size_t)k * 64 + cin) * 64 + cout]);
}

// ---------------- kernel 3: gather + WMMA conv, f32 out + deterministic BN partials --------------
// 1 wave (wave32) owns one 16-row tile and all 64 output channels (4 accumulators).
// A layout (16-bit, 16x32): lane<16 -> row=lane, elems 0..7 = K k0+0..7, elems 8..15 = K k0+16..23;
// lane>=16 -> row=lane-16, K k0+8..15 and k0+24..31.  -> four aligned 16B loads per row per offset.
// Streaming inputs (nbr_idx/nbr_valid, 216 MB once) use NT loads; the 256 MB conv output uses NT
// stores; this keeps the 128 MB x_bf gather working set resident in the 192 MB L2.
__global__ __launch_bounds__(128)
void conv_wmma(const unsigned short* __restrict__ xbf,
               const unsigned short* __restrict__ wfrag,
               const int* __restrict__ nidx,
               const int* __restrict__ nval,
               float* __restrict__ outbuf,
               float* __restrict__ partS,
               float* __restrict__ partQ)
{
  const int lane = threadIdx.x & 31;
  const int wave = threadIdx.x >> 5;
  const int tile = blockIdx.x * 4 + wave;              // 0..62499
  const int lrow = lane & 15;
  const int half = lane >> 4;
  const int row  = tile * 16 + lrow;

  v8f acc0 = {}, acc1 = {}, acc2 = {}, acc3 = {};
  const u32v4 zero4 = {0u, 0u, 0u, 0u};

  for (int k = 0; k < KOFF; ++k) {
    const int idx = __builtin_nontemporal_load(nidx + (size_t)k * NPTS + row);
    const int val = __builtin_nontemporal_load(nval + (size_t)k * NPTS + row);

    BfFrag A0, A1;                                     // K-chunks 0..31 and 32..63
    A0.u[0] = zero4; A0.u[1] = zero4; A1.u[0] = zero4; A1.u[1] = zero4;
    if (val) {                                         // skip gather entirely when invalid
      const u32v4* p = (const u32v4*)(xbf + (size_t)idx * CIN);
      A0.u[0] = p[half];        // K  0..7  / 8..15
      A0.u[1] = p[half + 2];    // K 16..23 / 24..31
      A1.u[0] = p[half + 4];    // K 32..39 / 40..47
      A1.u[1] = p[half + 6];    // K 48..55 / 56..63
    }

    const u32v4* wbase = (const u32v4*)wfrag + (size_t)k * 512 + lane * 2;  // 64 u32v4 per fragment
    BfFrag B;
    B.u[0] = wbase[0 * 64]; B.u[1] = wbase[0 * 64 + 1];
    acc0 = WMMA_BF16(A0.v, B.v, acc0);
    B.u[0] = wbase[1 * 64]; B.u[1] = wbase[1 * 64 + 1];
    acc1 = WMMA_BF16(A0.v, B.v, acc1);
    B.u[0] = wbase[2 * 64]; B.u[1] = wbase[2 * 64 + 1];
    acc2 = WMMA_BF16(A0.v, B.v, acc2);
    B.u[0] = wbase[3 * 64]; B.u[1] = wbase[3 * 64 + 1];
    acc3 = WMMA_BF16(A0.v, B.v, acc3);
    B.u[0] = wbase[4 * 64]; B.u[1] = wbase[4 * 64 + 1];
    acc0 = WMMA_BF16(A1.v, B.v, acc0);
    B.u[0] = wbase[5 * 64]; B.u[1] = wbase[5 * 64 + 1];
    acc1 = WMMA_BF16(A1.v, B.v, acc1);
    B.u[0] = wbase[6 * 64]; B.u[1] = wbase[6 * 64 + 1];
    acc2 = WMMA_BF16(A1.v, B.v, acc2);
    B.u[0] = wbase[7 * 64]; B.u[1] = wbase[7 * 64 + 1];
    acc3 = WMMA_BF16(A1.v, B.v, acc3);
  }

  // C/D layout: lane<16 -> col=lane rows 0..7 (VGPR r), lane>=16 -> same col rows 8..15.
  // Store conv result (bias cancels exactly through batch-norm, so it is omitted) and
  // emit per-tile per-channel sum / sum-of-squares partials (deterministic, no atomics).
#define EMIT_TILE(ACC, CT) do {                                             \
    const int c = (CT) * 16 + lrow;                                         \
    float s = 0.f, q = 0.f;                                                 \
    _Pragma("unroll")                                                       \
    for (int r = 0; r < 8; ++r) {                                           \
      float v = (ACC)[r];                                                   \
      __builtin_nontemporal_store(                                          \
          v, outbuf + (size_t)(tile * 16 + half * 8 + r) * COUT + c);       \
      s += v; q += v * v;                                                   \
    }                                                                       \
    s += __shfl_xor(s, 16, 32);                                             \
    q += __shfl_xor(q, 16, 32);                                             \
    if (half == 0) {                                                        \
      __builtin_nontemporal_store(s, partS + (size_t)tile * COUT + c);      \
      __builtin_nontemporal_store(q, partQ + (size_t)tile * COUT + c);      \
    }                                                                       \
  } while (0)

  EMIT_TILE(acc0, 0);
  EMIT_TILE(acc1, 1);
  EMIT_TILE(acc2, 2);
  EMIT_TILE(acc3, 3);
#undef EMIT_TILE
}

// ---------------- kernel 4: deterministic reduction -> per-channel scale/shift ----------------
__global__ __launch_bounds__(256)
void bn_stats(const float* __restrict__ partS, const float* __restrict__ partQ,
              const float* __restrict__ gamma, const float* __restrict__ beta,
              float* __restrict__ scsh)
{
  __shared__ float ss[256], qs[256];
  const int c = blockIdx.x;
  const int t = threadIdx.x;
  float s = 0.f, q = 0.f;
  for (int i = t; i < TILES; i += 256) {               // fixed strided order
    s += __builtin_nontemporal_load(partS + (size_t)i * COUT + c);
    q += __builtin_nontemporal_load(partQ + (size_t)i * COUT + c);
  }
  ss[t] = s; qs[t] = q;
  __syncthreads();
  for (int off = 128; off > 0; off >>= 1) {            // fixed tree order
    if (t < off) { ss[t] += ss[t + off]; qs[t] += qs[t + off]; }
    __syncthreads();
  }
  if (t == 0) {
    float mean = ss[0] / (float)NPTS;
    float var  = qs[0] / (float)NPTS - mean * mean;
    float sc   = gamma[c] * rsqrtf(var + 1e-5f);
    scsh[c]        = sc;
    scsh[COUT + c] = beta[c] - mean * sc;
  }
}

// ---------------- kernel 5: y = BN(out); out = y * sigmoid(y)   (in place on d_out) ----------------
__global__ __launch_bounds__(256)
void bn_silu(float* __restrict__ out, const float* __restrict__ scsh)
{
  __shared__ float sh[128];
  if (threadIdx.x < 128) sh[threadIdx.x] = scsh[threadIdx.x];
  __syncthreads();
  size_t base = ((size_t)blockIdx.x * 256 + threadIdx.x) * 4;   // N*64/4 threads
  int c = (int)(base & 63);
  f32x4 v = __builtin_nontemporal_load((const f32x4*)(out + base));
  float r[4] = {v.x, v.y, v.z, v.w};
#pragma unroll
  for (int j = 0; j < 4; ++j) {
    float y = r[j] * sh[c + j] + sh[64 + c + j];
    r[j] = y / (1.f + expf(-y));
  }
  f32x4 o = {r[0], r[1], r[2], r[3]};
  __builtin_nontemporal_store(o, (f32x4*)(out + base));
}

// ---------------- launcher ----------------
extern "C" void kernel_launch(void* const* d_in, const int* in_sizes, int n_in,
                              void* d_out, int out_size, void* d_ws, size_t ws_size,
                              hipStream_t stream)
{
  const float* feats = (const float*)d_in[0];
  const float* temb  = (const float*)d_in[1];
  const float* W     = (const float*)d_in[2];
  // d_in[3] = bias: cancels exactly under batch-norm (out-mean removes it) -> unused.
  const float* gamma = (const float*)d_in[4];
  const float* beta  = (const float*)d_in[5];
  const int* bidx    = (const int*)d_in[6];
  const int* nidx    = (const int*)d_in[7];
  const int* nval    = (const int*)d_in[8];

  char* ws = (char*)d_ws;
  unsigned short* xbf   = (unsigned short*)(ws);                 // 128,000,000 B
  unsigned short* wfrag = (unsigned short*)(ws + 128000000);     //     221,184 B
  float* partS          = (float*)(ws + 128221184);              //  16,000,000 B
  float* partQ          = (float*)(ws + 144221184);              //  16,000,000 B
  float* scsh           = (float*)(ws + 160221184);              //         512 B
  float* out            = (float*)d_out;                         // N*64 f32

  prep_x   <<<(NPTS * 16) / 256, 256, 0, stream>>>(feats, temb, bidx, xbf);
  prep_w   <<<(KOFF * 8 * 512) / 256, 256, 0, stream>>>(W, wfrag);
  conv_wmma<<<TILES / 4, 128, 0, stream>>>(xbf, wfrag, nidx, nval, out, partS, partQ);
  bn_stats <<<COUT, 256, 0, stream>>>(partS, partQ, gamma, beta, scsh);
  bn_silu  <<<(NPTS * COUT / 4) / 256, 256, 0, stream>>>(out, scsh);
}
